// MultiHeadSpatialNet_19430432047709
// MI455X (gfx1250) — compile-verified, matrix operand order checked by reading, not run on hardware
//
#include <hip/hip_runtime.h>
#include <hip/hip_bf16.h>
#include <math.h>

// ---------------------------------------------------------------------------
// MultiHeadSpatialNet on MI455X (gfx1250, wave32, WMMA).
// Score-fold + pool-before-project make this HBM-bound on the single 536 MB
// read of X; remaining dense math runs through v_wmma_f32_16x16x32_bf16 with
// f32 accumulation. Kernel 3 now uses 32 rows/block so each B-fragment load
// feeds two WMMA M-tiles (halves L2 weight traffic, fills WMMA hazard slots).
// ---------------------------------------------------------------------------

typedef __bf16 bf16_t;
typedef __attribute__((ext_vector_type(16))) __bf16 v16bf;
typedef __attribute__((ext_vector_type(8)))  __bf16 v8bf;
typedef __attribute__((ext_vector_type(8)))  float  v8f;

#define D_MODEL  512
#define N_TOK    64
#define N_HEAD   8
#define D_HIDDEN 2048
#define BATCH    4096
#define LN_EPS   1e-5f
#define ROWS     32     // batch rows per block in kernel 3 (2 M-tiles of 16)

// LDS row pitch (bf16 elems) for 512-wide tiles: 520*2B = 1040 B = 260 words,
// 260 % 64 = 4 -> the 16 A-fragment rows land on disjoint bank quartets.
#define PITCH 520

// ---- workspace layout (bytes) ----------------------------------------------
#define OFF_WKS   0                         // 8*512 f32            = 16 KB
#define OFF_WVSW  16384                     // 512 frags  *512 bf16 = 512 KB
#define OFF_W1SW  540672                    // 2048 frags *512 bf16 = 2 MB
#define OFF_W2SW  2637824                   // 2048 frags *512 bf16 = 2 MB
#define OFF_P     4734976                   // 4096*8*512 bf16      = 32 MB

static __device__ inline v16bf cat8(v8bf lo, v8bf hi) {
  return __builtin_shufflevector(lo, hi, 0,1,2,3,4,5,6,7,8,9,10,11,12,13,14,15);
}

static __device__ inline v16bf load_afrag(const bf16_t* p) {
  // Two aligned 16-B halves: K offsets {half*8+e} and {16+half*8+e} (ISA layout)
  return cat8(*(const v8bf*)p, *(const v8bf*)(p + 16));
}

static __device__ inline v8f wmma_bf16(v16bf a, v16bf b, v8f c) {
  // D = A(16x32 bf16) * B(32x16 bf16) + C(16x16 f32)
  return __builtin_amdgcn_wmma_f32_16x16x32_bf16(
      /*neg_a=*/false, a, /*neg_b=*/false, b,
      /*c_mod=*/(short)0, c, /*reuse_a=*/false, /*reuse_b=*/false);
}

// ---------------------------------------------------------------------------
// Kernel 1: fold Wk*score_w -> WkS[8][512]; swizzle Wv/W1/W2 into per-lane
// WMMA B-fragment order (lane<16: K in {0..7,16..23}; lane>=16: {8..15,24..31}).
// frag task f: [0,512)=Wv (32 ntile x 16 kstep), [512,2560)=W1 (128x16),
// [2560,4608)=W2 (32 ntile x 64 kstep). One lane-task writes 16 bf16.
// ---------------------------------------------------------------------------
__global__ __launch_bounds__(256) void prep_kernel(
    const float* __restrict__ Wk, const float* __restrict__ score_w,
    const float* __restrict__ Wv, const float* __restrict__ W1,
    const float* __restrict__ W2,
    float* __restrict__ WkS, bf16_t* __restrict__ WvSw,
    bf16_t* __restrict__ W1Sw, bf16_t* __restrict__ W2Sw) {
  const int NT_SWZ = 4608 * 32;
  const int NT_ALL = NT_SWZ + N_HEAD * D_MODEL;
  for (int t = blockIdx.x * blockDim.x + threadIdx.x; t < NT_ALL;
       t += gridDim.x * blockDim.x) {
    if (t < NT_SWZ) {
      const int f = t >> 5, lane = t & 31;
      const int half = lane >> 4, idx = lane & 15;
      const float* src; bf16_t* dst; int N, j, ks, fl;
      if (f < 512)       { fl = f;        src = Wv; dst = WvSw; N = 512;  j = fl >> 4; ks = fl & 15; }
      else if (f < 2560) { fl = f - 512;  src = W1; dst = W1Sw; N = 2048; j = fl >> 4; ks = fl & 15; }
      else               { fl = f - 2560; src = W2; dst = W2Sw; N = 512;  j = fl >> 6; ks = fl & 63; }
      const int n = j * 16 + idx;
      bf16_t* o = dst + ((size_t)fl * 32 + lane) * 16;
#pragma unroll
      for (int e = 0; e < 16; ++e) {
        const int k = ks * 32 + ((e < 8) ? (half * 8 + e) : (16 + half * 8 + (e - 8)));
        o[e] = (bf16_t)src[(size_t)k * N + n];
      }
    } else {
      const int t2 = t - NT_SWZ;
      const int h = t2 >> 9, m = t2 & 511;
      float acc = 0.f;
#pragma unroll 8
      for (int d = 0; d < 64; ++d)
        acc += Wk[(size_t)m * D_MODEL + h * 64 + d] * score_w[h * 64 + d];
      WkS[h * D_MODEL + m] = acc;  // [h][m], coalesced reads in kernel 2
    }
  }
}

// ---------------------------------------------------------------------------
// Kernel 2 (HBM-bound stage): one block per batch b. Streams the 128 KB X
// tile: folded scores -> softmax over N=64 -> pooled P[b,h,:] in bf16.
// bk / score_b omitted: constant over the softmax axis (shift invariance).
// Pool-phase X reads are non-temporal (last use; keep L2 for weights/P).
// ---------------------------------------------------------------------------
__global__ __launch_bounds__(256) void attn_pool_kernel(
    const float* __restrict__ X, const float* __restrict__ WkS,
    bf16_t* __restrict__ P) {
  __shared__ float s_sc[N_TOK][N_HEAD];
  __shared__ float s_at[N_TOK][N_HEAD];
  const int tid = threadIdx.x;
  const int b = blockIdx.x;
  const float* Xb = X + (size_t)b * N_TOK * D_MODEL;

  // Prefetch the whole tile (emits global_prefetch_b8); one cacheline per step.
  for (int p = tid; p < 1024; p += 256)
    __builtin_prefetch(Xb + p * 32, 0, 1);

  // scores: 512 (n,h) dots of length 512 against folded WkS
  for (int t = tid; t < N_TOK * N_HEAD; t += 256) {
    const int n = t >> 3, hh = t & 7;
    const float4* xr = (const float4*)(Xb + (size_t)n * D_MODEL);
    const float4* wr = (const float4*)(WkS + hh * D_MODEL);
    float acc = 0.f;
#pragma unroll 4
    for (int m4 = 0; m4 < D_MODEL / 4; ++m4) {
      const float4 xv = xr[m4], wv = wr[m4];
      acc += xv.x * wv.x + xv.y * wv.y + xv.z * wv.z + xv.w * wv.w;
    }
    s_sc[n][hh] = acc;
  }
  __syncthreads();

  // softmax over N per head (8 tiny serial columns)
  if (tid < N_HEAD) {
    float mx = -3.4e38f;
    for (int n = 0; n < N_TOK; ++n) mx = fmaxf(mx, s_sc[n][tid]);
    float sum = 0.f;
    for (int n = 0; n < N_TOK; ++n) {
      const float e = __expf(s_sc[n][tid] - mx);
      s_at[n][tid] = e; sum += e;
    }
    const float inv = 1.f / sum;
    for (int n = 0; n < N_TOK; ++n) s_at[n][tid] *= inv;
  }
  __syncthreads();

  // pooled P[h][m] = sum_n attn[n][h] * X[n][m]; X read once, coalesced in m.
  float acc[2][N_HEAD];
#pragma unroll
  for (int i = 0; i < 2; ++i)
#pragma unroll
    for (int hh = 0; hh < N_HEAD; ++hh) acc[i][hh] = 0.f;
  for (int n = 0; n < N_TOK; ++n) {
    const float xv0 = __builtin_nontemporal_load(Xb + (size_t)n * D_MODEL + tid);
    const float xv1 = __builtin_nontemporal_load(Xb + (size_t)n * D_MODEL + tid + 256);
#pragma unroll
    for (int hh = 0; hh < N_HEAD; ++hh) {
      const float a = s_at[n][hh];
      acc[0][hh] += a * xv0;
      acc[1][hh] += a * xv1;
    }
  }
#pragma unroll
  for (int hh = 0; hh < N_HEAD; ++hh) {
    P[((size_t)b * N_HEAD + hh) * D_MODEL + tid]       = (bf16_t)acc[0][hh];
    P[((size_t)b * N_HEAD + hh) * D_MODEL + tid + 256] = (bf16_t)acc[1][hh];
  }
}

// ---------------------------------------------------------------------------
// Kernel 3 (WMMA stage): 32 batch rows per block (2 M-tiles), 8 waves.
//   A: feature = per-head P @ Wv  (wave w = head w, 4 n-tiles, K=512)
//   B: FFN in 4 hidden chunks of 512: H=gelu(F@W1c) in LDS, Y += H@W2c
//      (Y lives in 64 f32 accumulator VGPRs per lane across all chunks)
//   C: +b2 +feature residual, LayerNorm via shfl_xor + LDS cross-wave reduce.
// Each B-fragment (one contiguous 32-B v16bf load from the pre-swizzled
// arrays) is reused by two WMMAs (M-tiles m=0,1) -> halves L2 weight traffic.
// ---------------------------------------------------------------------------
__global__ __launch_bounds__(256) void fused_ffn_kernel(
    const bf16_t* __restrict__ P, const bf16_t* __restrict__ WvSw,
    const bf16_t* __restrict__ W1Sw, const bf16_t* __restrict__ W2Sw,
    const float* __restrict__ bv, const float* __restrict__ b1,
    const float* __restrict__ b2, const float* __restrict__ ln_g,
    const float* __restrict__ ln_b, float* __restrict__ out) {
  __shared__ bf16_t bufA[ROWS * PITCH];   // feature (bf16), kept for residual
  __shared__ bf16_t bufH[ROWS * PITCH];   // gelu hidden chunk
  __shared__ float  s_red[ROWS][8][2];
  __shared__ float  s_mu[ROWS], s_rs[ROWS];

  const int tid  = threadIdx.x;
  const int wave = tid >> 5, lane = tid & 31;
  const int half = lane >> 4, idx = lane & 15;
  const int b0   = blockIdx.x * ROWS;

  // ---- Phase A: feature tiles (head h = wave) ------------------------------
  const int h = wave;
  const bf16_t* prow0 = P + ((size_t)(b0 + idx)      * N_HEAD + h) * D_MODEL + half * 8;
  const bf16_t* prow1 = P + ((size_t)(b0 + 16 + idx) * N_HEAD + h) * D_MODEL + half * 8;
#pragma unroll
  for (int t = 0; t < 4; ++t) {
    const int j = h * 4 + t;               // global n-tile in [0,32)
    v8f acc0 = {}, acc1 = {};
    for (int ks = 0; ks < 16; ++ks) {
      const v16bf bf = *(const v16bf*)(WvSw + ((size_t)(j * 16 + ks) * 32 + lane) * 16);
      acc0 = wmma_bf16(load_afrag(prow0 + ks * 32), bf, acc0);
      acc1 = wmma_bf16(load_afrag(prow1 + ks * 32), bf, acc1);
    }
#pragma unroll
    for (int r = 0; r < 8; ++r) {
      const int row = half * 8 + r, col = j * 16 + idx;
      bufA[row * PITCH + col]        = (bf16_t)(acc0[r] + bv[col]);
      bufA[(row + 16) * PITCH + col] = (bf16_t)(acc1[r] + bv[col]);
    }
  }
  __syncthreads();

  // ---- Phase B: FFN over 4 hidden chunks -----------------------------------
  v8f yacc[4][2] = {{{}, {}}, {{}, {}}, {{}, {}}, {{}, {}}};
  const bf16_t* arowA0 = bufA + idx * PITCH + half * 8;
  const bf16_t* arowA1 = bufA + (16 + idx) * PITCH + half * 8;
  const bf16_t* arowH0 = bufH + idx * PITCH + half * 8;
  const bf16_t* arowH1 = bufH + (16 + idx) * PITCH + half * 8;
  for (int c = 0; c < 4; ++c) {
    // H_chunk = gelu(F @ W1[:, c*512 .. c*512+511] + b1)
#pragma unroll
    for (int t = 0; t < 4; ++t) {
      const int j1 = c * 32 + wave * 4 + t;   // global W1 n-tile in [0,128)
      v8f acc0 = {}, acc1 = {};
      for (int ks = 0; ks < 16; ++ks) {
        const v16bf bf = *(const v16bf*)(W1Sw + ((size_t)(j1 * 16 + ks) * 32 + lane) * 16);
        acc0 = wmma_bf16(load_afrag(arowA0 + ks * 32), bf, acc0);
        acc1 = wmma_bf16(load_afrag(arowA1 + ks * 32), bf, acc1);
      }
#pragma unroll
      for (int r = 0; r < 8; ++r) {
        const int row = half * 8 + r;
        const int colg = j1 * 16 + idx;       // hidden col 0..2047
        const int coll = (j1 - c * 32) * 16 + idx;
        const float x0 = acc0[r] + b1[colg];
        const float x1 = acc1[r] + b1[colg];
        bufH[row * PITCH + coll] =
            (bf16_t)(0.5f * x0 * (1.f + erff(x0 * 0.70710678118654752f)));
        bufH[(row + 16) * PITCH + coll] =
            (bf16_t)(0.5f * x1 * (1.f + erff(x1 * 0.70710678118654752f)));
      }
    }
    __syncthreads();
    // Y += H_chunk @ W2[c*512 .. , :]
#pragma unroll
    for (int t = 0; t < 4; ++t) {
      const int jy = wave * 4 + t;            // output n-tile in [0,32)
      v8f acc0 = yacc[t][0], acc1 = yacc[t][1];
      for (int ks = 0; ks < 16; ++ks) {
        const int ksg = c * 16 + ks;          // global W2 kstep in [0,64)
        const v16bf bf = *(const v16bf*)(W2Sw + ((size_t)(jy * 64 + ksg) * 32 + lane) * 16);
        acc0 = wmma_bf16(load_afrag(arowH0 + ks * 32), bf, acc0);
        acc1 = wmma_bf16(load_afrag(arowH1 + ks * 32), bf, acc1);
      }
      yacc[t][0] = acc0; yacc[t][1] = acc1;
    }
    __syncthreads();   // bufH about to be overwritten by next chunk
  }

  // ---- Phase C: bias + residual + LayerNorm --------------------------------
  float vals[4][2][8];
#pragma unroll
  for (int t = 0; t < 4; ++t) {
    const int jy = wave * 4 + t;
#pragma unroll
    for (int m = 0; m < 2; ++m)
#pragma unroll
      for (int r = 0; r < 8; ++r) {
        const int row = m * 16 + half * 8 + r, col = jy * 16 + idx;
        vals[t][m][r] = yacc[t][m][r] + b2[col] + (float)bufA[row * PITCH + col];
      }
  }
  // per-row partials: reduce over the 16 lanes of this half (xor stays in-half)
#pragma unroll
  for (int m = 0; m < 2; ++m)
#pragma unroll
    for (int r = 0; r < 8; ++r) {
      float s = 0.f, q = 0.f;
#pragma unroll
      for (int t = 0; t < 4; ++t) { const float v = vals[t][m][r]; s += v; q += v * v; }
      for (int off = 1; off < 16; off <<= 1) {
        s += __shfl_xor(s, off, 32);
        q += __shfl_xor(q, off, 32);
      }
      if (idx == 0) {
        const int row = m * 16 + half * 8 + r;
        s_red[row][wave][0] = s;
        s_red[row][wave][1] = q;
      }
    }
  __syncthreads();
  if (tid < ROWS) {
    float s = 0.f, q = 0.f;
#pragma unroll
    for (int w = 0; w < 8; ++w) { s += s_red[tid][w][0]; q += s_red[tid][w][1]; }
    const float mu  = s * (1.f / D_MODEL);
    const float var = q * (1.f / D_MODEL) - mu * mu;
    s_mu[tid] = mu;
    s_rs[tid] = rsqrtf(var + LN_EPS);
  }
  __syncthreads();
#pragma unroll
  for (int t = 0; t < 4; ++t) {
    const int jy = wave * 4 + t;
#pragma unroll
    for (int m = 0; m < 2; ++m)
#pragma unroll
      for (int r = 0; r < 8; ++r) {
        const int row = m * 16 + half * 8 + r, col = jy * 16 + idx;
        __builtin_nontemporal_store(
            (vals[t][m][r] - s_mu[row]) * s_rs[row] * ln_g[col] + ln_b[col],
            out + (size_t)(b0 + row) * D_MODEL + col);
      }
  }
}

// ---------------------------------------------------------------------------
extern "C" void kernel_launch(void* const* d_in, const int* in_sizes, int n_in,
                              void* d_out, int out_size, void* d_ws, size_t ws_size,
                              hipStream_t stream) {
  const float* X       = (const float*)d_in[0];
  const float* Wk      = (const float*)d_in[1];
  // d_in[2] = bk      : cancels in softmax (shift invariance)
  const float* Wv      = (const float*)d_in[3];
  const float* bv      = (const float*)d_in[4];
  const float* score_w = (const float*)d_in[5];
  // d_in[6] = score_b : cancels in softmax
  const float* W1      = (const float*)d_in[7];
  const float* b1      = (const float*)d_in[8];
  const float* W2      = (const float*)d_in[9];
  const float* b2      = (const float*)d_in[10];
  const float* ln_g    = (const float*)d_in[11];
  const float* ln_b    = (const float*)d_in[12];
  float* out = (float*)d_out;

  char* ws = (char*)d_ws;
  float*  WkS  = (float*)(ws + OFF_WKS);
  bf16_t* WvSw = (bf16_t*)(ws + OFF_WVSW);
  bf16_t* W1Sw = (bf16_t*)(ws + OFF_W1SW);
  bf16_t* W2Sw = (bf16_t*)(ws + OFF_W2SW);
  bf16_t* P    = (bf16_t*)(ws + OFF_P);

  prep_kernel<<<592, 256, 0, stream>>>(Wk, score_w, Wv, W1, W2,
                                       WkS, WvSw, W1Sw, W2Sw);
  attn_pool_kernel<<<BATCH, 256, 0, stream>>>(X, WkS, P);
  fused_ffn_kernel<<<BATCH / ROWS, 256, 0, stream>>>(P, WvSw, W1Sw, W2Sw,
                                                     bv, b1, b2, ln_g, ln_b, out);
}
